// PELSTM_16475494548207
// MI455X (gfx1250) — matrix operand order repaired and checked by reading, not compiled, not run on hardware
//
#include <hip/hip_runtime.h>

// ---------------------------------------------------------------------------
// MI455X / gfx1250, wave32. bf16 WMMA (16x16x32, f32 accum) for the GEMM chain.
// M=64 token tiles per workgroup; each wave reuses every B-fragment across
// 4 M-tiles in registers -> 4x lower L2 traffic than an M=16 design.
// ---------------------------------------------------------------------------
typedef __bf16 bf16;
typedef bf16  v16bf __attribute__((ext_vector_type(16)));
typedef bf16  v8bf  __attribute__((ext_vector_type(8)));
typedef float v8f   __attribute__((ext_vector_type(8)));

#define TOKENS     (256 * 1024)
#define FRAG_HALF  512                        // elements (bf16) per 16x32x16 B-fragment
#define NFRAG      255                        // 52 + 91 + 91 + 21
#define FB_L1      0
#define FB_L2      52
#define FB_L3      143
#define FB_L4      234
#define XG0_BYTES  ((size_t)TOKENS * 40 * 4)  // 41,943,040
#define FRAG_BYTES ((size_t)NFRAG * FRAG_HALF * 2)
#define BIAS_FLOATS 672                       // 208 + 208 + 208 + 48
#define ASTR       232                        // LDS row stride (elems): 464B = 52 DW mod 64 banks -> conflict-free A loads

// ---------------------------------------------------------------------------
// Prep: pack weights into CDNA5 B-fragment layout (bf16).
// B 32x16: lane half h, elem e -> k = 8h + e (e<8), 16 + 8h + (e-8) (e>=8).
// B[k][n] = W[n][k]. Fragment index within a layer = nt*KSTEPS + ks;
// storage = frag*512 + lane*16 + e.
// ---------------------------------------------------------------------------
__global__ void pack_frags(const float* __restrict__ W1, const float* __restrict__ W2,
                           const float* __restrict__ W3, const float* __restrict__ Wih0,
                           bf16* __restrict__ frags) {
  int gid = blockIdx.x * blockDim.x + threadIdx.x;
  if (gid >= NFRAG * FRAG_HALF) return;
  int frag = gid >> 9;
  int within = gid & 511;
  int lane = within >> 4;
  int e = within & 15;
  const float* W; int Kin, Nout, nt, ks;
  if (frag < FB_L2)      { int f = frag;         nt = f / 4; ks = f % 4; W = W1;   Kin = 125; Nout = 200; }
  else if (frag < FB_L3) { int f = frag - FB_L2; nt = f / 7; ks = f % 7; W = W2;   Kin = 200; Nout = 200; }
  else if (frag < FB_L4) { int f = frag - FB_L3; nt = f / 7; ks = f % 7; W = W3;   Kin = 200; Nout = 196; }
  else                   { int f = frag - FB_L4; nt = f / 7; ks = f % 7; W = Wih0; Kin = 200; Nout = 40;  }
  int n = nt * 16 + (lane & 15);
  int k = ks * 32 + (e & 7) + ((e & 8) ? 16 : 0) + ((lane & 16) ? 8 : 0);
  float v = (n < Nout && k < Kin) ? W[n * Kin + k] : 0.f;
  frags[gid] = (bf16)v;
}

__global__ void pack_bias(const float* __restrict__ b1, const float* __restrict__ b2,
                          const float* __restrict__ b3, const float* __restrict__ bih0,
                          const float* __restrict__ bhh0, float* __restrict__ bias) {
  int gid = blockIdx.x * blockDim.x + threadIdx.x;
  if (gid >= BIAS_FLOATS) return;
  float v = 0.f;
  if (gid < 208)      { int n = gid;       if (n < 200) v = b1[n]; }
  else if (gid < 416) { int n = gid - 208; if (n < 200) v = b2[n]; }
  else if (gid < 624) { int n = gid - 416; if (n < 196) v = b3[n]; }
  else                { int n = gid - 624; if (n < 40)  v = bih0[n] + bhh0[n]; }
  bias[gid] = v;
}

// ---------------------------------------------------------------------------
// Workgroup GEMM layer: src (64 x ASTR bf16, LDS) x W -> relu -> dst (LDS).
// This wave handles NTW consecutive N-tiles starting at nt0, across all 4
// M-tiles. Each B-fragment (1KB from global/L2, WGP$-cached) feeds 4 WMMAs.
// A layout (16-bit A 16x32): lane<16: M=lane, e<8 -> k=e, e>=8 -> k=e+8;
// lane>=16: +8 on k -> two contiguous 16B LDS loads per fragment.
// C layout: lane gives col n=lane%16; vgpr r gives row r + (lane>=16 ? 8 : 0).
// ---------------------------------------------------------------------------
template <int NTW, int KS>
__device__ __forceinline__ void wg_gemm_relu(const bf16* __restrict__ src, bf16* __restrict__ dst,
                                             const bf16* __restrict__ fragL,
                                             const float* __restrict__ bias,
                                             int lane, int nt0) {
  v8f acc[4][NTW];
#pragma unroll
  for (int mt = 0; mt < 4; ++mt)
#pragma unroll
    for (int i = 0; i < NTW; ++i)
#pragma unroll
      for (int e = 0; e < 8; ++e) acc[mt][i][e] = 0.f;
  const int m = lane & 15;
  const int khi = (lane & 16) ? 8 : 0;
  const int rbase = (lane & 16) ? 8 : 0;
  for (int ks = 0; ks < KS; ++ks) {
    v16bf a[4];
#pragma unroll
    for (int mt = 0; mt < 4; ++mt) {
      const bf16* ap = src + (mt * 16 + m) * ASTR + ks * 32 + khi;
      v8bf a0 = *(const v8bf*)(ap);
      v8bf a1 = *(const v8bf*)(ap + 16);
#pragma unroll
      for (int e = 0; e < 8; ++e) { a[mt][e] = a0[e]; a[mt][e + 8] = a1[e]; }
    }
#pragma unroll
    for (int i = 0; i < NTW; ++i) {
      const bf16* bp = fragL + ((nt0 + i) * KS + ks) * FRAG_HALF + lane * 16;
      v8bf b0 = *(const v8bf*)(bp);
      v8bf b1 = *(const v8bf*)(bp + 8);
      v16bf bb;
#pragma unroll
      for (int e = 0; e < 8; ++e) { bb[e] = b0[e]; bb[e + 8] = b1[e]; }
#pragma unroll
      for (int mt = 0; mt < 4; ++mt)
        acc[mt][i] = __builtin_amdgcn_wmma_f32_16x16x32_bf16(false, a[mt], false, bb,
                                                             (short)0, acc[mt][i], false, false);
    }
  }
#pragma unroll
  for (int i = 0; i < NTW; ++i) {
    int n = (nt0 + i) * 16 + m;
    float bv = bias[n];
#pragma unroll
    for (int mt = 0; mt < 4; ++mt)
#pragma unroll
      for (int r = 0; r < 8; ++r) {
        float v = acc[mt][i][r] + bv;
        dst[(mt * 16 + rbase + r) * ASTR + n] = (bf16)(v > 0.f ? v : 0.f);
      }
  }
}

// Final layer (Wih0): one N-tile for this wave, 4 M-tiles; store fp32 gate
// preactivations to xg0[token][40].
template <int KS>
__device__ __forceinline__ void wg_gemm_store(const bf16* __restrict__ src,
                                              const bf16* __restrict__ fragL,
                                              const float* __restrict__ bias, int lane,
                                              int nt0, float* __restrict__ xg0, long tok0) {
  v8f acc[4];
#pragma unroll
  for (int mt = 0; mt < 4; ++mt)
#pragma unroll
    for (int e = 0; e < 8; ++e) acc[mt][e] = 0.f;
  const int m = lane & 15;
  const int khi = (lane & 16) ? 8 : 0;
  const int rbase = (lane & 16) ? 8 : 0;
  for (int ks = 0; ks < KS; ++ks) {
    v16bf a[4];
#pragma unroll
    for (int mt = 0; mt < 4; ++mt) {
      const bf16* ap = src + (mt * 16 + m) * ASTR + ks * 32 + khi;
      v8bf a0 = *(const v8bf*)(ap);
      v8bf a1 = *(const v8bf*)(ap + 16);
#pragma unroll
      for (int e = 0; e < 8; ++e) { a[mt][e] = a0[e]; a[mt][e + 8] = a1[e]; }
    }
    const bf16* bp = fragL + (nt0 * KS + ks) * FRAG_HALF + lane * 16;
    v8bf b0 = *(const v8bf*)(bp);
    v8bf b1 = *(const v8bf*)(bp + 8);
    v16bf bb;
#pragma unroll
    for (int e = 0; e < 8; ++e) { bb[e] = b0[e]; bb[e + 8] = b1[e]; }
#pragma unroll
    for (int mt = 0; mt < 4; ++mt)
      acc[mt] = __builtin_amdgcn_wmma_f32_16x16x32_bf16(false, a[mt], false, bb,
                                                        (short)0, acc[mt], false, false);
  }
  int n = nt0 * 16 + m;
  if (n < 40) {
    float bv = bias[n];
#pragma unroll
    for (int mt = 0; mt < 4; ++mt)
#pragma unroll
      for (int r = 0; r < 8; ++r) {
        long t = tok0 + mt * 16 + rbase + r;
        xg0[t * 40 + n] = acc[mt][r] + bv;
      }
  }
}

// ---------------------------------------------------------------------------
// Fused MLP: one workgroup (4 waves) per 64 tokens. Activations ping-pong in
// shared LDS (2 x 64 x 232 bf16 = 58KB); N-tiles split 4/3/3/3 across waves.
// Whole 125->200->200->196(+x2)->40 chain stays on-chip.
// ---------------------------------------------------------------------------
__global__ __launch_bounds__(128) void mlp_kernel(const float* __restrict__ x1,
                                                  const float* __restrict__ x2,
                                                  const bf16* __restrict__ frags,
                                                  const float* __restrict__ bias,
                                                  float* __restrict__ xg0) {
  __shared__ __align__(16) bf16 act[2][64 * ASTR];  // 59,392 B
  const int tid = threadIdx.x;
  const int w = tid >> 5;
  const int lane = tid & 31;
  const long tok0 = (long)blockIdx.x * 64;

  // zero both buffers (padding cols 200..231 must stay 0 for the K=224 layers)
  for (int i = tid; i < 64 * ASTR; i += 128) {
    act[0][i] = (bf16)0.f;
    act[1][i] = (bf16)0.f;
  }
  // stage input tile: concat(x1[121], x2[4]) -> cols 0..124
  for (int r = w; r < 64; r += 4) {
    long t = tok0 + r;
    for (int c = lane; c < 125; c += 32) {
      float v = (c < 121) ? x1[t * 121 + c] : x2[t * 4 + (c - 121)];
      act[0][r * ASTR + c] = (bf16)v;
    }
  }
  __syncthreads();

  const int nt0 = (w == 0) ? 0 : (w == 1) ? 4 : (w == 2) ? 7 : 10;  // 4/3/3/3 split of 13 tiles

  if (w == 0) wg_gemm_relu<4, 4>(act[0], act[1], frags + FB_L1 * FRAG_HALF, bias + 0, lane, nt0);
  else        wg_gemm_relu<3, 4>(act[0], act[1], frags + FB_L1 * FRAG_HALF, bias + 0, lane, nt0);
  __syncthreads();

  if (w == 0) wg_gemm_relu<4, 7>(act[1], act[0], frags + FB_L2 * FRAG_HALF, bias + 208, lane, nt0);
  else        wg_gemm_relu<3, 7>(act[1], act[0], frags + FB_L2 * FRAG_HALF, bias + 208, lane, nt0);
  __syncthreads();

  if (w == 0) wg_gemm_relu<4, 7>(act[0], act[1], frags + FB_L3 * FRAG_HALF, bias + 416, lane, nt0);
  else        wg_gemm_relu<3, 7>(act[0], act[1], frags + FB_L3 * FRAG_HALF, bias + 416, lane, nt0);
  __syncthreads();

  // append x2 into cols 196..199 (LSTM input = concat(a3, x2) = 200 wide)
  for (int i = tid; i < 64 * 4; i += 128) {
    int r = i >> 2, c = i & 3;
    act[1][r * ASTR + 196 + c] = (bf16)x2[(tok0 + r) * 4 + c];
  }
  __syncthreads();

  // final 200->40 projection: waves 0..2 take N-tiles 0..2 (40 cols)
  if (w < 3)
    wg_gemm_store<7>(act[1], frags + FB_L4 * FRAG_HALF, bias + 624, lane, w, xg0, tok0);
}

// ---------------------------------------------------------------------------
// Sequential LSTM scan. 3 batch elems / wave (lanes 0..29 = (batch,unit)).
// h/c in registers; h broadcast across the 10 unit-lanes via __shfl
// (-> ds_bpermute, wave-synchronous, no barriers). Wih1@h computed inline.
// ---------------------------------------------------------------------------
__device__ __forceinline__ float sigm(float x) { return 1.f / (1.f + __expf(-x)); }
__device__ __forceinline__ float tanh_f(float x) {
  x = fminf(15.f, fmaxf(-15.f, x));
  float e = __expf(2.f * x);
  return (e - 1.f) / (e + 1.f);
}

__global__ __launch_bounds__(128) void lstm_kernel(
    const float* __restrict__ xg0, const float* __restrict__ Whh0,
    const float* __restrict__ Wih1, const float* __restrict__ Whh1,
    const float* __restrict__ bih1, const float* __restrict__ bhh1,
    const float* __restrict__ W4, const float* __restrict__ b4,
    float* __restrict__ out) {
  const int lane = threadIdx.x & 31;
  const int wave = blockIdx.x * (blockDim.x >> 5) + (threadIdx.x >> 5);
  const int blw = lane / 10;          // batch-in-wave 0..2 (lanes 30,31 idle)
  const int j = lane - blw * 10;      // hidden unit 0..9
  const int b = wave * 3 + blw;
  const bool act = (lane < 30) && (b < 256);

  float whh0[4][10], wih1[4][10], whh1[4][10], bias1[4], w4[10];
#pragma unroll
  for (int gi = 0; gi < 4; ++gi) {
    int row = gi * 10 + j;
    bias1[gi] = act ? (bih1[row] + bhh1[row]) : 0.f;
#pragma unroll
    for (int k = 0; k < 10; ++k) {
      whh0[gi][k] = act ? Whh0[row * 10 + k] : 0.f;
      wih1[gi][k] = act ? Wih1[row * 10 + k] : 0.f;
      whh1[gi][k] = act ? Whh1[row * 10 + k] : 0.f;
    }
  }
#pragma unroll
  for (int k = 0; k < 10; ++k) w4[k] = W4[k];

  const float* xg = xg0 + (size_t)(act ? b : 0) * 1024 * 40;
  float h0 = 0.f, c0 = 0.f, h1 = 0.f, c1 = 0.f;

  for (int s = 0; s < 1024; ++s) {
    float hv[10];
#pragma unroll
    for (int k = 0; k < 10; ++k) hv[k] = __shfl(h0, blw * 10 + k, 32);
    float g[4];
#pragma unroll
    for (int gi = 0; gi < 4; ++gi) {
      float a = act ? xg[s * 40 + gi * 10 + j] : 0.f;
#pragma unroll
      for (int k = 0; k < 10; ++k) a += whh0[gi][k] * hv[k];
      g[gi] = a;
    }
    c0 = sigm(g[1]) * c0 + sigm(g[0]) * tanh_f(g[2]);
    h0 = sigm(g[3]) * tanh_f(c0);

    float h0v[10], h1v[10];
#pragma unroll
    for (int k = 0; k < 10; ++k) {
      h0v[k] = __shfl(h0, blw * 10 + k, 32);
      h1v[k] = __shfl(h1, blw * 10 + k, 32);
    }
#pragma unroll
    for (int gi = 0; gi < 4; ++gi) {
      float a = bias1[gi];
#pragma unroll
      for (int k = 0; k < 10; ++k) a += wih1[gi][k] * h0v[k] + whh1[gi][k] * h1v[k];
      g[gi] = a;
    }
    c1 = sigm(g[1]) * c1 + sigm(g[0]) * tanh_f(g[2]);
    h1 = sigm(g[3]) * tanh_f(c1);
  }

  float h1v[10];
#pragma unroll
  for (int k = 0; k < 10; ++k) h1v[k] = __shfl(h1, blw * 10 + k, 32);
  if (act && j == 0) {
    float o = b4[0];
#pragma unroll
    for (int k = 0; k < 10; ++k) o += w4[k] * h1v[k];
    out[b] = o;
  }
}

// ---------------------------------------------------------------------------
// Host launcher. d_in order: x1,x2,W1,b1,W2,b2,W3,b3,Wih0,Whh0,bih0,bhh0,
//                            Wih1,Whh1,bih1,bhh1,W4,b4
// ws layout: [xg0: 41,943,040 B][frags: 261,120 B][bias: 2,688 B]
// ---------------------------------------------------------------------------
extern "C" void kernel_launch(void* const* d_in, const int* in_sizes, int n_in,
                              void* d_out, int out_size, void* d_ws, size_t ws_size,
                              hipStream_t stream) {
  const float* x1   = (const float*)d_in[0];
  const float* x2   = (const float*)d_in[1];
  const float* W1   = (const float*)d_in[2];
  const float* b1   = (const float*)d_in[3];
  const float* W2   = (const float*)d_in[4];
  const float* b2   = (const float*)d_in[5];
  const float* W3   = (const float*)d_in[6];
  const float* b3   = (const float*)d_in[7];
  const float* Wih0 = (const float*)d_in[8];
  const float* Whh0 = (const float*)d_in[9];
  const float* bih0 = (const float*)d_in[10];
  const float* bhh0 = (const float*)d_in[11];
  const float* Wih1 = (const float*)d_in[12];
  const float* Whh1 = (const float*)d_in[13];
  const float* bih1 = (const float*)d_in[14];
  const float* bhh1 = (const float*)d_in[15];
  const float* W4   = (const float*)d_in[16];
  const float* b4   = (const float*)d_in[17];

  char* ws = (char*)d_ws;
  float* xg0   = (float*)ws;
  bf16*  frags = (bf16*)(ws + XG0_BYTES);
  float* biasp = (float*)(ws + XG0_BYTES + FRAG_BYTES);

  pack_frags<<<(NFRAG * FRAG_HALF + 255) / 256, 256, 0, stream>>>(W1, W2, W3, Wih0, frags);
  pack_bias<<<3, 256, 0, stream>>>(b1, b2, b3, bih0, bhh0, biasp);
  mlp_kernel<<<TOKENS / 64, 128, 0, stream>>>(x1, x2, frags, biasp, xg0);
  lstm_kernel<<<22, 128, 0, stream>>>(xg0, Whh0, Wih1, Whh1, bih1, bhh1, W4, b4,
                                      (float*)d_out);
}
